// NISER_ODE_68478958567755
// MI455X (gfx1250) — matrix operand order repaired and checked by reading, not compiled, not run on hardware
//
#include <hip/hip_runtime.h>
#include <math.h>

#define NN 65536
#define EE 131072
#define BB 4096
#define NPG 16
#define DD 128
#define VV 50000
#define ESESS 32
#define NPB 391   // number of 128-wide column tiles covering V (391*128 >= 50000)

typedef __attribute__((ext_vector_type(16))) _Float16 v16h;
typedef __attribute__((ext_vector_type(8)))  float    v8f;
typedef _Float16 h16;

__device__ __forceinline__ float sigmoidf_(float x){ return 1.0f/(1.0f+expf(-x)); }

__device__ __forceinline__ v8f wmma_f16(v16h a, v16h b, v8f c){
  return __builtin_amdgcn_wmma_f32_16x16x32_f16(false, a, false, b, (short)0, c, false, false);
}

// A-fragment (16x32 f16, contiguous-k layout) from row-major f16 tile with ld=128.
// lane<16: M=lane, K 0..15; lane>=16: M=lane-16, K 16..31
__device__ __forceinline__ v16h load_afrag(const h16* tile, int lane, int kt){
  int m  = lane & 15;
  int c0 = (lane < 16) ? 0 : 16;
  const h16* p = tile + m*DD + kt*32 + c0;
  union { v16h v; float4 f[2]; } u;
  u.f[0] = *(const float4*)(p);
  u.f[1] = *(const float4*)(p + 8);
  return u.v;
}
__device__ __forceinline__ v16h load_afrag_g(const h16* base, int row0, int lane, int kt){
  int m  = lane & 15;
  int c0 = (lane < 16) ? 0 : 16;
  const h16* p = base + (size_t)(row0 + m)*DD + kt*32 + c0;
  union { v16h v; float4 f[2]; } u;
  u.f[0] = *(const float4*)(p);
  u.f[1] = *(const float4*)(p + 8);
  return u.v;
}
// B-fragment (32x16 f16) where B[k][n] = baseT[n][k] (baseT row-major [.,128], e.g. tgt16).
// lane<16: N=lane, K=kt*32+0..15 ; lane>=16: N=lane-16, K=kt*32+16..31 -> one contiguous 32B load
__device__ __forceinline__ v16h load_bfrag_T(const h16* baseT, int col0, int lane, int kt){
  int n  = lane & 15;
  int k0 = kt*32 + ((lane < 16) ? 0 : 16);
  return *(const v16h*)(baseT + (size_t)(col0 + n)*DD + k0);
}

// ---------------- t_end = max(edge_t) ----------------
__global__ void k_tmax_init(unsigned* p){ if (threadIdx.x==0 && blockIdx.x==0) p[0]=0u; }

__global__ __launch_bounds__(256) void k_tmax(const float* __restrict__ et, unsigned* __restrict__ p){
  int idx = blockIdx.x*256 + threadIdx.x;
  float m = 0.0f;
  for (int e = idx; e < EE; e += gridDim.x*256) m = fmaxf(m, et[e]);
  for (int o = 16; o > 0; o >>= 1) m = fmaxf(m, __shfl_xor(m, o, 32));
  if ((threadIdx.x & 31) == 0) atomicMax(p, __float_as_uint(m));  // all values >= 0
}

// ---------------- weight prep: combine + f16 + WMMA-B swizzle ----------------
// wsw[w][((kt*8+nt)*32+lane)*16 + i] = W[k][n], k=kt*32+(lane<16?0:16)+i, n=nt*16+(lane&15)
__global__ __launch_bounds__(256) void k_prep_w(
    const float* __restrict__ Wxr, const float* __restrict__ Whr,
    const float* __restrict__ Wxz, const float* __restrict__ Whz,
    const float* __restrict__ Wxh, const float* __restrict__ Whh,
    const float* __restrict__ Wu,  h16* __restrict__ wsw)
{
  int o = blockIdx.x*256 + threadIdx.x;        // 0 .. 5*16384-1
  int wi = o >> 14;
  int p  = o & 16383;
  int i = p & 15, lane = (p>>4)&31, nt = (p>>9)&7, kt = (p>>12)&3;
  int k = kt*32 + ((lane < 16) ? 0 : 16) + i;
  int n = nt*16 + (lane & 15);
  int idx = k*DD + n;
  float v;
  if      (wi == 0) v = Wxr[idx] + Whr[idx];
  else if (wi == 1) v = Wxz[idx] + Whz[idx];
  else if (wi == 2) v = Wxh[idx];
  else if (wi == 3) v = Whh[idx];
  else              v = Wu[idx];
  wsw[o] = (h16)v;
}

__global__ __launch_bounds__(128) void k_prep_b(
    const float* __restrict__ bxr, const float* __restrict__ bhr,
    const float* __restrict__ bxz, const float* __restrict__ bhz,
    float* __restrict__ br, float* __restrict__ bz)
{
  int t = threadIdx.x;
  br[t] = bxr[t] + bhr[t];
  bz[t] = bxz[t] + bhz[t];
}

// ---------------- tgt = normalize(emb) -> f16 ----------------
__global__ __launch_bounds__(32) void k_prep_tgt(const float* __restrict__ emb, h16* __restrict__ tgt16){
  const int v = blockIdx.x;
  const int t = threadIdx.x;
  float4 x = *(const float4*)(emb + (size_t)v*DD + t*4);
  float s = x.x*x.x + x.y*x.y + x.z*x.z + x.w*x.w;
  for (int o = 16; o > 0; o >>= 1) s += __shfl_xor(s, o, 32);
  float inv = 1.0f / (sqrtf(s) + 1e-12f);
  h16* op = tgt16 + (size_t)v*DD + t*4;
  op[0] = (h16)(x.x*inv); op[1] = (h16)(x.y*inv);
  op[2] = (h16)(x.z*inv); op[3] = (h16)(x.w*inv);
}

// ---------------- fused per-session kernel (1 wave / session) ----------------
__global__ __launch_bounds__(32) void k_session(
    const int* __restrict__ iid, const int* __restrict__ srcg, const int* __restrict__ dstg,
    const float* __restrict__ edge_t, const float* __restrict__ emb,
    const h16* __restrict__ wsw,
    const float* __restrict__ br, const float* __restrict__ bz,
    const float* __restrict__ bxh, const float* __restrict__ bhh,
    const float* __restrict__ Wv, const float* __restrict__ bv,
    const float* __restrict__ We, const float* __restrict__ Wsr,
    const float* __restrict__ tendp,
    h16* __restrict__ sr16)
{
  __shared__ __align__(16) float feat[NPG][DD];
  __shared__ __align__(16) h16   a16[NPG][DD];
  __shared__ __align__(16) float rbuf[NPG][DD];
  __shared__ __align__(16) float zbuf[NPG][DD];
  __shared__ __align__(16) float xh[NPG][DD];      // pre-act xh, later feat_u
  __shared__ __align__(16) float agg[NPG][DD];
  __shared__ float no_s[NPG], ni_s[NPG];
  __shared__ float fv[DD];
  __shared__ float srg[DD];
  __shared__ int   esl[ESESS], edl[ESESS];
  __shared__ float ew[ESESS];
  __shared__ float scb[NPG];
  __shared__ float alpha[NPG];

  const int g = blockIdx.x;
  const int t = threadIdx.x;
  const float t1 = tendp[0] * 0.1f;
  const int col = t & 15;
  const int mbase = (t >> 4) << 3;   // 0 or 8 (C-fragment row base)

  // 1) feat = L2-normalize(emb[iid])
  for (int r = 0; r < NPG; ++r){
    const float* er = emb + (size_t)iid[g*NPG + r] * DD;
    float4 v = *(const float4*)(er + t*4);
    float s = v.x*v.x + v.y*v.y + v.z*v.z + v.w*v.w;
    for (int o = 16; o > 0; o >>= 1) s += __shfl_xor(s, o, 32);
    float inv = 1.0f / (sqrtf(s) + 1e-12f);
    *(float4*)(&feat[r][t*4]) = make_float4(v.x*inv, v.y*inv, v.z*inv, v.w*inv);
  }

  // 2) edges, active-edge filter, degree norms
  {
    int e  = g*ESESS + t;
    int sl = srcg[e] - g*NPG;
    int dl = dstg[e] - g*NPG;
    float w = (edge_t[e] <= 0.0f) ? 1.0f : 0.0f;
    esl[t] = sl; edl[t] = dl; ew[t] = w;
    if (t < NPG){ no_s[t] = 0.0f; ni_s[t] = 0.0f; }
    __syncthreads();
    atomicAdd(&no_s[sl], w);
    atomicAdd(&ni_s[dl], w);
    __syncthreads();
    if (t < NPG){
      no_s[t] = rsqrtf(fmaxf(no_s[t], 1.0f));
      ni_s[t] = rsqrtf(fmaxf(ni_s[t], 1.0f));
    }
    __syncthreads();
  }

  // 3) agg(feat); A = agg * ni  (f16)
  for (int j = 0; j < 4; ++j){ int d = t*4+j; for (int r = 0; r < NPG; ++r) agg[r][d] = 0.0f; }
  __syncthreads();
  for (int e = 0; e < ESESS; ++e){
    if (ew[e] > 0.0f){
      int sl = esl[e], dl = edl[e];
      float sf = no_s[sl];
      for (int j = 0; j < 4; ++j){ int d = t*4+j; agg[dl][d] += feat[sl][d] * sf; }
    }
  }
  __syncthreads();
  for (int j = 0; j < 4; ++j){ int d = t*4+j;
    for (int r = 0; r < NPG; ++r) a16[r][d] = (h16)(agg[r][d] * ni_s[r]); }
  __syncthreads();

  v16h af[4];
  for (int kt = 0; kt < 4; ++kt) af[kt] = load_afrag(&a16[0][0], t, kt);

  const h16* Wr  = wsw;
  const h16* Wz  = wsw + 16384;
  const h16* Wx  = wsw + 2*16384;
  const h16* Wh2 = wsw + 3*16384;
  const h16* Wu  = wsw + 4*16384;

  // 4) r, z, xh pre-activations: A @ {Wr,Wz,Wxh} + bias
  for (int nt = 0; nt < 8; ++nt){
    int c = nt*16 + col;
    v8f ar, az, ax;
    float b0 = br[c], b1 = bz[c], b2 = bxh[c];
    for (int i = 0; i < 8; ++i){ ar[i] = b0; az[i] = b1; ax[i] = b2; }
    for (int kt = 0; kt < 4; ++kt){
      size_t fo = (size_t)((kt*8 + nt)*32 + t) * 16;
      ar = wmma_f16(af[kt], *(const v16h*)(Wr + fo), ar);
      az = wmma_f16(af[kt], *(const v16h*)(Wz + fo), az);
      ax = wmma_f16(af[kt], *(const v16h*)(Wx + fo), ax);
    }
    for (int i = 0; i < 8; ++i){
      int m = mbase + i;
      rbuf[m][c] = sigmoidf_(ar[i]);
      zbuf[m][c] = sigmoidf_(az[i]);
      xh[m][c]   = ax[i];
    }
  }
  __syncthreads();

  // 5) agg(r*h); A2 = agg * ni
  for (int j = 0; j < 4; ++j){ int d = t*4+j; for (int r = 0; r < NPG; ++r) agg[r][d] = 0.0f; }
  __syncthreads();
  for (int e = 0; e < ESESS; ++e){
    if (ew[e] > 0.0f){
      int sl = esl[e], dl = edl[e];
      float sf = no_s[sl];
      for (int j = 0; j < 4; ++j){ int d = t*4+j; agg[dl][d] += rbuf[sl][d]*feat[sl][d]*sf; }
    }
  }
  __syncthreads();
  for (int j = 0; j < 4; ++j){ int d = t*4+j;
    for (int r = 0; r < NPG; ++r) a16[r][d] = (h16)(agg[r][d] * ni_s[r]); }
  __syncthreads();

  v16h af2[4];
  for (int kt = 0; kt < 4; ++kt) af2[kt] = load_afrag(&a16[0][0], t, kt);

  // 6) u = tanh(xh + A2@Whh + bhh); Euler update
  for (int nt = 0; nt < 8; ++nt){
    int c = nt*16 + col;
    v8f acc; float b0 = bhh[c];
    for (int i = 0; i < 8; ++i) acc[i] = b0;
    for (int kt = 0; kt < 4; ++kt)
      acc = wmma_f16(af2[kt], *(const v16h*)(Wh2 + (size_t)((kt*8 + nt)*32 + t)*16), acc);
    for (int i = 0; i < 8; ++i){
      int m = mbase + i;
      float u_ = tanhf(acc[i] + xh[m][c]);
      float h  = feat[m][c];
      float z  = zbuf[m][c];
      feat[m][c] = h + t1 * (1.0f - z) * (u_ - h);
    }
  }
  __syncthreads();

  // 7) renormalize (no eps)
  for (int r = 0; r < NPG; ++r){
    float4 v = *(const float4*)(&feat[r][t*4]);
    float s = v.x*v.x + v.y*v.y + v.z*v.z + v.w*v.w;
    for (int o = 16; o > 0; o >>= 1) s += __shfl_xor(s, o, 32);
    float inv = 1.0f / sqrtf(s);
    *(float4*)(&feat[r][t*4]) = make_float4(v.x*inv, v.y*inv, v.z*inv, v.w*inv);
  }
  __syncthreads();

  // 8) feat_u = feat @ Wu -> xh
  for (int j = 0; j < 4; ++j){ int d = t*4+j;
    for (int r = 0; r < NPG; ++r) a16[r][d] = (h16)feat[r][d]; }
  __syncthreads();
  v16h af3[4];
  for (int kt = 0; kt < 4; ++kt) af3[kt] = load_afrag(&a16[0][0], t, kt);
  for (int nt = 0; nt < 8; ++nt){
    int c = nt*16 + col;
    v8f acc = {};
    for (int kt = 0; kt < 4; ++kt)
      acc = wmma_f16(af3[kt], *(const v16h*)(Wu + (size_t)((kt*8 + nt)*32 + t)*16), acc);
    for (int i = 0; i < 8; ++i) xh[mbase + i][c] = acc[i];
  }
  __syncthreads();

  // 9) feat_v = feat[last] @ Wv + bv  (scalar, tiny)
  for (int j = 0; j < 4; ++j){
    int d = t*4+j;
    float s = bv[d];
    for (int k = 0; k < DD; ++k) s += feat[NPG-1][k] * Wv[k*DD + d];
    fv[d] = s;
  }
  __syncthreads();

  // 10) attention scores + segment softmax (16 nodes)
  for (int r = 0; r < NPG; ++r){
    float s = 0.0f;
    for (int j = 0; j < 4; ++j){ int d = t*4+j; s += sigmoidf_(xh[r][d] + fv[d]) * We[d]; }
    for (int o = 16; o > 0; o >>= 1) s += __shfl_xor(s, o, 32);
    if (t == 0) scb[r] = s;
  }
  __syncthreads();
  float emax = -1e30f;
  for (int r = 0; r < NPG; ++r) emax = fmaxf(emax, scb[r]);
  if (t < NPG) alpha[t] = expf(scb[t] - emax);
  __syncthreads();
  float den = 0.0f;
  for (int r = 0; r < NPG; ++r) den += alpha[r];
  float dinv = 1.0f / den;

  // 11) sr_g
  for (int j = 0; j < 4; ++j){
    int d = t*4+j;
    float s = 0.0f;
    for (int r = 0; r < NPG; ++r) s += feat[r][d] * alpha[r];
    srg[d] = s * dinv;
  }
  __syncthreads();

  // 12) sr = [sr_l, sr_g] @ Wsr; normalize; fold SCALE=12 into f16 output
  float vo[4]; float ss = 0.0f;
  for (int j = 0; j < 4; ++j){
    int c = t*4+j;
    float s = 0.0f;
    for (int k = 0; k < DD; ++k) s += feat[NPG-1][k] * Wsr[k*DD + c];
    for (int k = 0; k < DD; ++k) s += srg[k] * Wsr[(DD + k)*DD + c];
    vo[j] = s; ss += s*s;
  }
  for (int o = 16; o > 0; o >>= 1) ss += __shfl_xor(ss, o, 32);
  float inv = 12.0f / (sqrtf(ss) + 1e-12f);
  for (int j = 0; j < 4; ++j)
    sr16[(size_t)g*DD + t*4 + j] = (h16)(vo[j] * inv);
}

// ---------------- logits = sr16 @ tgt16^T (x12 folded) + fused exp-sum partials ----
// 1 wave / block, 4 M-tiles (64 rows) x 128 cols. |logit| <= 12 so exp is safe
// without max subtraction. Partials written once per (row, col-block): deterministic.
__global__ __launch_bounds__(32) void k_logits(
    const h16* __restrict__ sr16, const h16* __restrict__ tgt16,
    float* __restrict__ out, float* __restrict__ pw)
{
  const int t  = threadIdx.x;
  const int bx = blockIdx.x;
  const int m0 = blockIdx.y * 64;
  const int c0 = bx * 128;
  const int col = t & 15;
  const int mbase = (t >> 4) << 3;

  v16h af[4][4];
  for (int mt = 0; mt < 4; ++mt)
    for (int kt = 0; kt < 4; ++kt)
      af[mt][kt] = load_afrag_g(sr16, m0 + mt*16, t, kt);

  float es0[8], es1[8], es2[8], es3[8];
  for (int i = 0; i < 8; ++i){ es0[i]=0.0f; es1[i]=0.0f; es2[i]=0.0f; es3[i]=0.0f; }

  for (int nt = 0; nt < 8; ++nt){
    int ct = c0 + nt*16;
    if (ct >= VV) break;                 // uniform (V % 16 == 0)
    v8f a0 = {}, a1 = {}, a2 = {}, a3 = {};
    for (int kt = 0; kt < 4; ++kt){
      v16h bf = load_bfrag_T(tgt16, ct, t, kt);
      a0 = wmma_f16(af[0][kt], bf, a0);
      a1 = wmma_f16(af[1][kt], bf, a1);
      a2 = wmma_f16(af[2][kt], bf, a2);
      a3 = wmma_f16(af[3][kt], bf, a3);
    }
    int c = ct + col;
    for (int i = 0; i < 8; ++i){
      int m = mbase + i;
      float x0 = a0[i], x1 = a1[i], x2 = a2[i], x3 = a3[i];
      out[(size_t)(m0 +  0 + m)*VV + c] = x0;  es0[i] += expf(x0);
      out[(size_t)(m0 + 16 + m)*VV + c] = x1;  es1[i] += expf(x1);
      out[(size_t)(m0 + 32 + m)*VV + c] = x2;  es2[i] += expf(x2);
      out[(size_t)(m0 + 48 + m)*VV + c] = x3;  es3[i] += expf(x3);
    }
  }

  // reduce exp-sums across the 16 lanes of each C-fragment half-group
  for (int mt = 0; mt < 4; ++mt){
    for (int i = 0; i < 8; ++i){
      float v = (mt==0) ? es0[i] : (mt==1) ? es1[i] : (mt==2) ? es2[i] : es3[i];
      v += __shfl_xor(v, 1, 32);
      v += __shfl_xor(v, 2, 32);
      v += __shfl_xor(v, 4, 32);
      v += __shfl_xor(v, 8, 32);
      if (col == 0)
        pw[(size_t)(m0 + mt*16 + mbase + i)*NPB + bx] = v;
    }
  }
}

// ---------------- per-row lse = log(sum of partials), fixed order ----------------
__global__ __launch_bounds__(256) void k_rowlse(const float* __restrict__ pw, float* __restrict__ lse){
  int r = blockIdx.x*256 + threadIdx.x;          // 0 .. BB-1
  if (r >= BB) return;
  const float* p = pw + (size_t)r*NPB;
  float s = 0.0f;
  for (int j = 0; j < NPB; ++j) s += p[j];
  lse[r] = logf(s);
}

// ---------------- out -= lse[row] (vectorized, V % 4 == 0) ----------------
__global__ __launch_bounds__(256) void k_sub(float* __restrict__ out, const float* __restrict__ lse){
  const size_t n4 = (size_t)BB * VV / 4;
  float4* o4 = (float4*)out;
  for (size_t i = (size_t)blockIdx.x*blockDim.x + threadIdx.x; i < n4;
       i += (size_t)gridDim.x*blockDim.x){
    float4 v = o4[i];
    float l = lse[(i*4) / VV];                   // 4 elems never straddle a row
    v.x -= l; v.y -= l; v.z -= l; v.w -= l;
    o4[i] = v;
  }
}

extern "C" void kernel_launch(void* const* d_in, const int* in_sizes, int n_in,
                              void* d_out, int out_size, void* d_ws, size_t ws_size,
                              hipStream_t stream) {
  const int*   iid    = (const int*)  d_in[0];
  const int*   src    = (const int*)  d_in[1];
  const int*   dst    = (const int*)  d_in[2];
  const float* edge_t = (const float*)d_in[3];
  const float* emb    = (const float*)d_in[6];
  const float* Wxr = (const float*)d_in[7];  const float* bxr = (const float*)d_in[8];
  const float* Wxz = (const float*)d_in[9];  const float* bxz = (const float*)d_in[10];
  const float* Wxh = (const float*)d_in[11]; const float* bxh = (const float*)d_in[12];
  const float* Whr = (const float*)d_in[13]; const float* bhr = (const float*)d_in[14];
  const float* Whz = (const float*)d_in[15]; const float* bhz = (const float*)d_in[16];
  const float* Whh = (const float*)d_in[17]; const float* bhh = (const float*)d_in[18];
  const float* Wu  = (const float*)d_in[19]; const float* Wv  = (const float*)d_in[20];
  const float* bv  = (const float*)d_in[21]; const float* We  = (const float*)d_in[22];
  const float* Wsr = (const float*)d_in[23];
  float* out = (float*)d_out;

  char* ws = (char*)d_ws;
  unsigned* tend = (unsigned*)ws;                                   // 4B (padded to 256)
  h16*   wsw  = (h16*)(ws + 256);                                   // 5*16384 f16
  float* brc  = (float*)(ws + 256 + 5*16384*2);                     // 128 f32
  float* bzc  = brc + DD;                                           // 128 f32
  h16*   tgt16 = (h16*)(ws + 256 + 5*16384*2 + 1024);               // V*128 f16
  h16*   sr16  = tgt16 + (size_t)VV*DD;                             // B*128 f16
  float* pw    = (float*)((char*)(sr16 + (size_t)BB*DD));           // B*NPB f32
  float* lse   = pw + (size_t)BB*NPB;                               // B f32

  k_tmax_init<<<1, 1, 0, stream>>>(tend);
  k_tmax<<<128, 256, 0, stream>>>(edge_t, tend);
  k_prep_w<<<(5*16384)/256, 256, 0, stream>>>(Wxr, Whr, Wxz, Whz, Wxh, Whh, Wu, wsw);
  k_prep_b<<<1, 128, 0, stream>>>(bxr, bhr, bxz, bhz, brc, bzc);
  k_prep_tgt<<<VV, 32, 0, stream>>>(emb, tgt16);
  k_session<<<BB, 32, 0, stream>>>(iid, src, dst, edge_t, emb, wsw,
                                   brc, bzc, bxh, bhh, Wv, bv, We, Wsr,
                                   (const float*)tend, sr16);
  k_logits<<<dim3(NPB, BB/64), 32, 0, stream>>>(sr16, tgt16, out, pw);
  k_rowlse<<<(BB + 255)/256, 256, 0, stream>>>(pw, lse);
  k_sub<<<8192, 256, 0, stream>>>(out, lse);
}
